// OuterProductMean_63574105915472
// MI455X (gfx1250) — matrix operand order repaired and checked by reading, not compiled, not run on hardware
//
#include <hip/hip_runtime.h>
#include <cstddef>
#include <cstdint>

// ---------------- shapes ----------------
#define MSA   256      // m
#define NSEQ  384      // n (i, j)
#define DM    256      // dim_msa
#define DH    32       // hidden per side
#define DP    128      // dim_pairwise
#define CD    (DH*DH)     // 1024 (K of second GEMM)
#define EPS   1e-5f

typedef __attribute__((ext_vector_type(16))) _Float16 v16h;
typedef __attribute__((ext_vector_type(8)))  float    v8f;

union FragH { v16h v; _Float16 h[16]; unsigned int u[8]; };
union FragF { v8f  v; float f[8]; };

// gfx1250 async-to-LDS path (guarded: falls back to plain copies if absent)
#if defined(__HIP_DEVICE_COMPILE__) && \
    __has_builtin(__builtin_amdgcn_global_load_async_to_lds_b128) && \
    __has_builtin(__builtin_amdgcn_s_wait_asynccnt)
#define USE_ASYNC 1
#define ASYNC_WAIT(n) __builtin_amdgcn_s_wait_asynccnt(n)
typedef int vi4 __attribute__((vector_size(16)));          // matches builtin param
typedef __attribute__((address_space(1))) vi4 gvi4;        // global int4
typedef __attribute__((address_space(3))) vi4 lvi4;        // LDS int4
#else
#define USE_ASYNC 0
#define ASYNC_WAIT(n)
#endif

// A-operand (16xK f16), ISA 7.12.2: lanes 0-15 & 16-31 both hold M=0..15;
// source LDS/global layout is [row][K] with K contiguous -> b128-mergeable.
static __device__ __forceinline__ v16h load_a_frag(const _Float16* base, int row,
                                                   int stride_h, int lane) {
  FragH f;
  const _Float16* p = base + (size_t)row * stride_h + ((lane & 16) ? 16 : 0);
#pragma unroll
  for (int v = 0; v < 8; ++v) {
    int K = 2 * v;
    f.u[v] = *(const unsigned int*)(p + K);
  }
  return f.v;
}

// B-operand (Kx16 f16): lane = N column; lanes 0-15 K=0..15, lanes 16-31 K=16..31
static __device__ __forceinline__ v16h load_b_frag(const _Float16* base, int row,
                                                   int stride_h, int lane) {
  FragH f;
  const _Float16* p = base + (size_t)row * stride_h + ((lane & 16) ? 16 : 0);
#pragma unroll
  for (int v = 0; v < 8; ++v)
    f.u[v] = *(const unsigned int*)(p + 2 * v);
  return f.v;
}

// ---- K1: LayerNorm + projections, emitting TRANSPOSED [n*dh][m] f16 outputs --
// Block = 8 waves covering 8 MSA rows (m0..m0+7) of the SAME column i, so the
// transposed store is 16B-contiguous along m.
__global__ void __launch_bounds__(256)
opm_ln_proj(const float* __restrict__ x, const unsigned char* __restrict__ mask,
            const float* __restrict__ g, const float* __restrict__ b,
            const float* __restrict__ wl, const float* __restrict__ bl,
            const float* __restrict__ wr, const float* __restrict__ br,
            _Float16* __restrict__ LfT, _Float16* __restrict__ RfT,
            float* __restrict__ maskf) {
  __shared__ float xs[8][DM];                       // 8 KB
  __shared__ __align__(16) _Float16 lt[DH][8];      // [c][wave(m)]
  __shared__ __align__(16) _Float16 rt[DH][8];
  const int lane = threadIdx.x & 31;
  const int w    = threadIdx.x >> 5;
  const int i    = blockIdx.x >> 5;                 // 0..383
  const int m0   = (blockIdx.x & 31) * 8;           // 0..248
  const int row  = (m0 + w) * NSEQ + i;             // into x / mask

  const float* xr = x + (size_t)row * DM;
  float vals[8], s = 0.f, s2 = 0.f;
#pragma unroll
  for (int q = 0; q < 8; ++q) {
    float v = xr[lane + q * 32];
    vals[q] = v; s += v; s2 += v * v;
  }
#pragma unroll
  for (int off = 16; off > 0; off >>= 1) {          // wave32 reduction
    s  += __shfl_xor(s,  off);
    s2 += __shfl_xor(s2, off);
  }
  const float mu   = s  * (1.f / DM);
  const float var  = s2 * (1.f / DM) - mu * mu;
  const float rstd = rsqrtf(var + EPS);
#pragma unroll
  for (int q = 0; q < 8; ++q) {
    int k = lane + q * 32;
    xs[w][k] = (vals[q] - mu) * rstd * g[k] + b[k];
  }
  __syncthreads();

  const float mf = mask[row] ? 1.f : 0.f;           // lane-uniform
  float la = bl[lane], ra = br[lane];               // lane = c in [0,32)
  for (int k = 0; k < DM; ++k) {
    float xv = xs[w][k];
    la = fmaf(xv, wl[k * DH + lane], la);
    ra = fmaf(xv, wr[k * DH + lane], ra);
  }
  lt[lane][w] = (_Float16)(la * mf);
  rt[lane][w] = (_Float16)(ra * mf);
  if (lane == 0) maskf[row] = mf;
  __syncthreads();

  if (threadIdx.x < 64) {                           // 16B row stores: [i*32+c][m0..m0+7]
    int  c   = threadIdx.x & 31;
    bool isR = threadIdx.x >= 32;
    const _Float16* src = isR ? &rt[c][0] : &lt[c][0];
    _Float16* dst = (isR ? RfT : LfT) + (size_t)(i * DH + c) * MSA + m0;
    *(uint4*)dst = *(const uint4*)src;
  }
}

// ---- K2: wo (1024x128 f32) -> wo_t (128x1024 f16) ---------------------------
__global__ void __launch_bounds__(256)
opm_wo_prep(const float* __restrict__ wo, _Float16* __restrict__ wo_t) {
  int idx = blockIdx.x * 256 + threadIdx.x;         // 131072
  int k = idx >> 7, p = idx & 127;
  wo_t[(size_t)p * CD + k] = (_Float16)wo[idx];
}

// ---- K3: invc[i,j] = 1/(sum_m mf[m,i]*mf[m,j] + eps) ------------------------
__global__ void __launch_bounds__(256)
opm_count(const float* __restrict__ maskf, float* __restrict__ invc) {
  int idx = blockIdx.x * 256 + threadIdx.x;         // 147456
  int i = idx / NSEQ, j = idx - i * NSEQ;
  float c = 0.f;
  for (int m = 0; m < MSA; ++m)
    c = fmaf(maskf[m * NSEQ + i], maskf[m * NSEQ + j], c);
  invc[idx] = 1.f / (c + EPS);
}

// ---- K4: fused outer-product-sum + wo projection ----------------------------
// 8 waves; tile = 4i x 4j = 16 pairs. Phase A: C(128x128) over K=m=256 with
// double-buffered async LDS tiles. Phase B: out(16x128) = Cvec(16x1024) @ wo.
__global__ void __launch_bounds__(256)
opm_outer(const _Float16* __restrict__ LfT, const _Float16* __restrict__ RfT,
          const _Float16* __restrict__ wo_t, const float* __restrict__ invc,
          const float* __restrict__ bo, float* __restrict__ out) {
  __shared__ __align__(16) _Float16 smem[16384];    // 32 KB; 2x(Lt+Rt) / Cvec
  _Float16* Cvec = smem;                            // [16][1024] in phase B

  const int t    = threadIdx.x;
  const int lane = t & 31;
  const int w    = t >> 5;
  const int i0   = blockIdx.y * 4;
  const int j0   = blockIdx.x * 4;

  v8f acc[8] = {};

  // stage one 16(k) x 256(rows) f16 tile pair into buffer `buf` ([mm][kk])
  auto issue_tile = [&](int k0, int buf) {
#pragma unroll
    for (int r = 0; r < 2; ++r) {
      int task = t + r * 256;                       // 512 tasks of 8 halves
      int mm = task >> 1, seg = (task & 1) * 8;
      const _Float16* gl = LfT + (size_t)(i0 * DH + mm) * MSA + k0 + seg;
      const _Float16* gr = RfT + (size_t)(j0 * DH + mm) * MSA + k0 + seg;
      _Float16* ll = smem + buf * 8192 + mm * 32 + seg;
      _Float16* lr = smem + buf * 8192 + 4096 + mm * 32 + seg;
#if USE_ASYNC
      __builtin_amdgcn_global_load_async_to_lds_b128((gvi4*)gl, (lvi4*)ll, 0, 0);
      __builtin_amdgcn_global_load_async_to_lds_b128((gvi4*)gr, (lvi4*)lr, 0, 0);
#else
      *(uint4*)ll = *(const uint4*)gl;
      *(uint4*)lr = *(const uint4*)gr;
#endif
    }
  };

  issue_tile(0, 0);
#pragma unroll
  for (int kt = 0; kt < 8; ++kt) {
    if (kt + 1 < 8) {                               // prefetch next tile pair
      issue_tile((kt + 1) * 32, (kt + 1) & 1);
      ASYNC_WAIT(4);                                // current tile's 4 DMAs done
    } else {
      ASYNC_WAIT(0);
    }
    __syncthreads();

    const _Float16* Lt = smem + (kt & 1) * 8192;
    const _Float16* Rt = Lt + 4096;
    v16h af = load_a_frag(Lt, w * 16 + (lane & 15), 32, lane);
#pragma unroll
    for (int nt = 0; nt < 8; ++nt) {
      v16h bf = load_b_frag(Rt, nt * 16 + (lane & 15), 32, lane);
      acc[nt] = __builtin_amdgcn_wmma_f32_16x16x32_f16(
          false, af, false, bf, (short)0, acc[nt], false, false);
    }
    __syncthreads();                                // buffer reuse fence
  }

  // scatter C (f32 VGPRs) -> Cvec[pair][c*32+d] as f16
#pragma unroll
  for (int nt = 0; nt < 8; ++nt) {
    FragF c; c.v = acc[nt];
#pragma unroll
    for (int r = 0; r < 8; ++r) {
      int M = w * 16 + r + ((lane & 16) ? 8 : 0);   // (i_local, c)
      int N = nt * 16 + (lane & 15);                // (j_local, d)
      int pair = ((M >> 5) << 2) + (N >> 5);
      int cd   = (M & 31) * 32 + (N & 31);
      Cvec[pair * CD + cd] = (_Float16)c.f[r];
    }
  }
  __syncthreads();

  // Phase B: out(16 pairs x 128) = Cvec @ wo ; wave w covers p = w*16..w*16+15
  v8f oacc = {};
  for (int k0 = 0; k0 < CD; k0 += 32) {
    v16h av = load_a_frag(Cvec + k0, lane & 15, CD, lane);
    v16h bv = load_b_frag(wo_t + k0, w * 16 + (lane & 15), CD, lane);
    oacc = __builtin_amdgcn_wmma_f32_16x16x32_f16(
        false, av, false, bv, (short)0, oacc, false, false);
  }

  FragF o; o.v = oacc;
  const int p = w * 16 + (lane & 15);
  const float bop = bo[p];
#pragma unroll
  for (int r = 0; r < 8; ++r) {
    int pair = r + ((lane & 16) ? 8 : 0);
    int i = i0 + (pair >> 2), j = j0 + (pair & 3);
    float ic = invc[i * NSEQ + j];
    out[((size_t)i * NSEQ + j) * DP + p] = (o.f[r] + bop) * ic;
  }
}

// ---------------- launch -----------------------------------------------------
extern "C" void kernel_launch(void* const* d_in, const int* in_sizes, int n_in,
                              void* d_out, int out_size, void* d_ws, size_t ws_size,
                              hipStream_t stream) {
  const float*          x    = (const float*)d_in[0];
  const unsigned char*  mask = (const unsigned char*)d_in[1];
  const float*          ln_g = (const float*)d_in[2];
  const float*          ln_b = (const float*)d_in[3];
  const float*          wl   = (const float*)d_in[4];
  const float*          bl   = (const float*)d_in[5];
  const float*          wr   = (const float*)d_in[6];
  const float*          br   = (const float*)d_in[7];
  const float*          wo   = (const float*)d_in[8];
  const float*          bo   = (const float*)d_in[9];
  float* out = (float*)d_out;

  // workspace layout (16B-aligned sections, ~13.2 MB total)
  char* ws = (char*)d_ws;
  _Float16* LfT   = (_Float16*)(ws + 0);             // [12288][256] f16 = 6 MB
  _Float16* RfT   = (_Float16*)(ws + 6291456);       // 6 MB
  float*    maskf = (float*)   (ws + 12582912);      // 98304 f32
  _Float16* wo_t  = (_Float16*)(ws + 12976128);      // 128*1024 f16
  float*    invc  = (float*)   (ws + 13238272);      // 147456 f32

  opm_ln_proj<<<dim3(12288), dim3(256), 0, stream>>>(x, mask, ln_g, ln_b,
                                                     wl, bl, wr, br, LfT, RfT, maskf);
  opm_wo_prep<<<dim3(512), dim3(256), 0, stream>>>(wo, wo_t);
  opm_count  <<<dim3(576), dim3(256), 0, stream>>>(maskf, invc);
  opm_outer  <<<dim3(96, 96), dim3(256), 0, stream>>>(LfT, RfT, wo_t, invc, bo, out);
}
